// GRU_87771951661159
// MI455X (gfx1250) — compile-verified, matrix operand order checked by reading, not run on hardware
//
#include <hip/hip_runtime.h>
#include <hip/hip_bf16.h>

// ---------------------------------------------------------------------------
// GRU on MI455X (gfx1250, wave32, WMMA + TDM):
//   proj   = x @ Wp.T + bp                (bf16 WMMA GEMM, f32 accum)
//   scan   = per-head recurrent WMMA loop; gate inputs streamed one step
//            ahead into an LDS ring by the Tensor Data Mover (TENSORcnt)
//   out    = states @ Wout.T              (bf16 WMMA GEMM)
// ---------------------------------------------------------------------------

typedef __attribute__((ext_vector_type(16))) __bf16 v16bf;
typedef __attribute__((ext_vector_type(8)))  float  v8f;
typedef __attribute__((ext_vector_type(4)))  unsigned int v4u;
typedef __attribute__((ext_vector_type(8)))  int         v8i;
typedef __attribute__((ext_vector_type(4)))  int         v4i;

union Frag16 { unsigned u[8]; v16bf v; };   // 16x32 (A) or 32x16 (B) bf16 frag
union AccF   { float    f[8]; v8f   v; };   // 16x16 f32 C/D frag

static __device__ __forceinline__ unsigned pack2bf(float lo, float hi) {
    unsigned short ul = __builtin_bit_cast(unsigned short, (__bf16)lo);
    unsigned short uh = __builtin_bit_cast(unsigned short, (__bf16)hi);
    return ((unsigned)uh << 16) | (unsigned)ul;
}

static __device__ __forceinline__ float sigmoidf_fast(float x) {
    return 1.0f / (1.0f + __expf(-x));
}

#if defined(__has_builtin)
#if __has_builtin(__builtin_amdgcn_tensor_load_to_lds) && \
    __has_builtin(__builtin_amdgcn_s_wait_tensorcnt)
#define USE_TDM 1
#endif
#endif
#ifndef USE_TDM
#define USE_TDM 0
#endif

// ---------------------------------------------------------------------------
// Kernel 1/3: C[M,Ncols] = A[M,K] @ W[Ncols,K]^T (+ bias), f32 in/out,
// bf16 WMMA inner product. Block tile 128x128, BK=32, 8 waves (2M x 4N),
// each wave: 4 (M) x 2 (N) 16x16 accumulators.
// ---------------------------------------------------------------------------
__global__ __launch_bounds__(256)
void gemm_bf16_wmma(const float* __restrict__ A, const float* __restrict__ W,
                    const float* __restrict__ bias, float* __restrict__ C,
                    int M, int K, int Ncols)
{
    __shared__ __bf16 As[128][34];   // [m][k], padded rows: conflict-free frag reads
    __shared__ __bf16 Ws[128][34];   // [n][k]

    const int tid   = threadIdx.x;
    const int lane  = tid & 31;
    const int wid   = tid >> 5;
    const int waveM = wid >> 2;            // 0..1 -> 64 rows
    const int waveN = wid & 3;             // 0..3 -> 32 cols
    const int bm0   = blockIdx.y * 128;
    const int bn0   = blockIdx.x * 128;

    const int lm = lane & 15;
    const int ks = (lane & 16) ? 8 : 0;

    AccF acc[4][2];
    #pragma unroll
    for (int mt = 0; mt < 4; ++mt)
        #pragma unroll
        for (int nt = 0; nt < 2; ++nt)
            #pragma unroll
            for (int i = 0; i < 8; ++i) acc[mt][nt].f[i] = 0.0f;

    const int lrow = tid >> 3;             // 0..31
    const int lcol = (tid & 7) * 4;        // 0,4,...,28

    for (int kk = 0; kk < K; kk += 32) {
        __syncthreads();
        #pragma unroll
        for (int r = 0; r < 4; ++r) {
            const int row = lrow + 32 * r;
            const float4 av = *(const float4*)&A[(size_t)(bm0 + row) * K + kk + lcol];
            As[row][lcol + 0] = (__bf16)av.x;
            As[row][lcol + 1] = (__bf16)av.y;
            As[row][lcol + 2] = (__bf16)av.z;
            As[row][lcol + 3] = (__bf16)av.w;
            const float4 wv = *(const float4*)&W[(size_t)(bn0 + row) * K + kk + lcol];
            Ws[row][lcol + 0] = (__bf16)wv.x;
            Ws[row][lcol + 1] = (__bf16)wv.y;
            Ws[row][lcol + 2] = (__bf16)wv.z;
            Ws[row][lcol + 3] = (__bf16)wv.w;
            if (kk + 32 < K) {   // hint next K tile toward the caches
                __builtin_prefetch(&A[(size_t)(bm0 + row) * K + kk + 32 + lcol], 0, 1);
                __builtin_prefetch(&W[(size_t)(bn0 + row) * K + kk + 32 + lcol], 0, 1);
            }
        }
        __syncthreads();

        Frag16 bfr[2];
        #pragma unroll
        for (int nt = 0; nt < 2; ++nt) {
            const int n = waveN * 32 + nt * 16 + lm;
            #pragma unroll
            for (int v = 0; v < 8; ++v) {
                const int k = (v < 4) ? (ks + 2 * v) : (16 + ks + 2 * (v - 4));
                bfr[nt].u[v] = *(const unsigned*)&Ws[n][k];
            }
        }
        #pragma unroll
        for (int mt = 0; mt < 4; ++mt) {
            Frag16 afr;
            const int m = waveM * 64 + mt * 16 + lm;
            #pragma unroll
            for (int v = 0; v < 8; ++v) {
                const int k = (v < 4) ? (ks + 2 * v) : (16 + ks + 2 * (v - 4));
                afr.u[v] = *(const unsigned*)&As[m][k];
            }
            #pragma unroll
            for (int nt = 0; nt < 2; ++nt)
                acc[mt][nt].v = __builtin_amdgcn_wmma_f32_16x16x32_bf16(
                    false, afr.v, false, bfr[nt].v, (short)0, acc[mt][nt].v,
                    false, false);
        }
    }

    const int rbase = (lane & 16) ? 8 : 0;
    #pragma unroll
    for (int mt = 0; mt < 4; ++mt)
        #pragma unroll
        for (int nt = 0; nt < 2; ++nt)
            #pragma unroll
            for (int v = 0; v < 8; ++v) {
                const int gm = bm0 + waveM * 64 + mt * 16 + rbase + v;
                const int gn = bn0 + waveN * 32 + nt * 16 + lm;
                float val = acc[mt][nt].f[v];
                if (bias) val += bias[gn];
                C[(size_t)gm * Ncols + gn] = val;
            }
}

// ---------------------------------------------------------------------------
// TDM helper: issue one tensor_load_to_lds moving the (4 batch x 3 gate x
// 128 col) f32 gate slice for timestep t into an LDS ring slot.
//   x: 128 contiguous floats          (tile_dim0=128, tensor_dim0_stride=1024)
//   y: 3 gates, stride 1024 floats    (tile_dim1=3)
//   z: 4 batches, stride S*3072       (tile_dim2=4, tensor_dim1_stride)
// This toolchain exposes the 6-arg builtin: 4 D# SGPR groups + an extended
// (zero-filled) dword group + cache-policy immediate.
// ---------------------------------------------------------------------------
#if USE_TDM
static __device__ __forceinline__ void tdm_issue_gate_slice(
    const float* gsrc /* tile start: proj + head*128 + t*3072 */,
    void* lds_dst)
{
    const unsigned long long ga = (unsigned long long)(const void*)gsrc;
    const unsigned lds_addr = (unsigned)(unsigned long long)lds_dst;

    const unsigned long long bstride = 2048ull * 3072ull;   // S*3*1024 elements

    v4u g0;
    g0[0] = 1u;                                        // count=1, valid user D#
    g0[1] = lds_addr;                                  // lds_addr (bytes)
    g0[2] = (unsigned)(ga & 0xFFFFFFFFu);              // global_addr[31:0]
    g0[3] = (unsigned)((ga >> 32) & 0x01FFFFFFu)       // global_addr[56:32]
          | (2u << 30);                                // type = 2 ("image")

    v8i g1;
    g1[0] = (int)(2u << 16);                           // data_size=2 (4 bytes)
    g1[1] = (int)(128u << 16);                         // tensor_dim0 = 128 (lo16)
    g1[2] = (int)(3u << 16);                           // dim0 hi=0 | tensor_dim1=3
    g1[3] = (int)(128u << 16);                         // dim1 hi=0 | tile_dim0=128
    g1[4] = (int)(3u | (4u << 16));                    // tile_dim1=3 | tile_dim2=4
    g1[5] = (int)1024;                                 // tensor_dim0_stride lo32
    g1[6] = (int)(((unsigned)(bstride & 0xFFFFu)) << 16); // stride0 hi | stride1 lo16
    g1[7] = (int)(bstride >> 16);                      // tensor_dim1_stride [47:16]

    v4i g2;
    g2[0] = 4;                                         // tensor_dim2 = 4
    g2[1] = 0;                                         // tensor_dim3 / lds_inc
    g2[2] = 0;                                         // tensor_dim2_stride lo
    g2[3] = 0;                                         // hi | tile_dim3=0

    v4i g3 = {0, 0, 0, 0};
    v8i g4 = {0, 0, 0, 0, 0, 0, 0, 0};                 // extended group (unused)

    __builtin_amdgcn_tensor_load_to_lds(g0, g1, g2, g3, g4, 0);
}
#endif

// ---------------------------------------------------------------------------
// Kernel 2: recurrent scan. One block per head (8 blocks, 256 threads).
// Wave w owns output columns [16w,16w+16) of H=128; weight B-frags (3 gates x
// 4 K-chunks) live in VGPRs for all 2048 steps. h exchanged as packed bf16
// pairs in LDS so A-frags are 8 ds_load_b32 each. Gate inputs are streamed
// one step ahead by the TDM into a depth-2 LDS ring (s_wait_tensorcnt before
// the end-of-step barrier publishes the slot to all waves).
// ---------------------------------------------------------------------------
__global__ __launch_bounds__(256)
void gru_scan_wmma(const float* __restrict__ proj,   // (B,S,3*1024)
                   const float* __restrict__ Wc,     // state  (N,H,H)
                   const float* __restrict__ Wf,     // forget (N,H,H)
                   const float* __restrict__ Wr,     // reset  (N,H,H)
                   float* __restrict__ states,       // (B,S,1024)
                   float* __restrict__ h_last)       // (B,1024)
{
    constexpr int S = 2048, H = 128, B = 4;

    __shared__ __bf16 h_bf[B][H];    // current hidden state (bf16)
    __shared__ __bf16 hr_bf[B][H];   // h * reset_gate (bf16)
#if USE_TDM
    __shared__ float  ring[2][B * 3 * H];   // gate-input ring, 2 x 6 KB
#endif

    const int head = blockIdx.x;
    const int tid  = threadIdx.x;
    const int lane = tid & 31;
    const int wid  = tid >> 5;                 // 0..7 -> 16-col slice
    const int lm   = lane & 15;
    const int ks   = (lane & 16) ? 8 : 0;
    const bool act = (lane < 16);
    const int ncol = wid * 16 + lm;            // column within head
    const int doff = head * H + ncol;          // column within D_STATE

    // ---- one-time: load weight B-fragments (contraction dim = W row) ----
    Frag16 fWc[4], fWf[4], fWr[4];
    const float* wcp = Wc + (size_t)head * H * H;
    const float* wfp = Wf + (size_t)head * H * H;
    const float* wrp = Wr + (size_t)head * H * H;
    #pragma unroll
    for (int c = 0; c < 4; ++c)
        #pragma unroll
        for (int v = 0; v < 8; ++v) {
            const int k = c * 32 + ((v < 4) ? (ks + 2 * v) : (16 + ks + 2 * (v - 4)));
            fWc[c].u[v] = pack2bf(wcp[(size_t)k * H + ncol], wcp[(size_t)(k + 1) * H + ncol]);
            fWf[c].u[v] = pack2bf(wfp[(size_t)k * H + ncol], wfp[(size_t)(k + 1) * H + ncol]);
            fWr[c].u[v] = pack2bf(wrp[(size_t)k * H + ncol], wrp[(size_t)(k + 1) * H + ncol]);
        }

    for (int i = tid; i < B * H; i += 256) ((__bf16*)h_bf)[i] = (__bf16)0.0f;
    float hreg[B] = {0.f, 0.f, 0.f, 0.f};      // lane-private h for own column

    const size_t pstride_b = (size_t)S * 3072;
    const float* gate_base = proj + (size_t)head * H;   // + t*3072 per step

#if USE_TDM
    if (wid == 0) {                            // prime slot 0 with t=0
        tdm_issue_gate_slice(gate_base, &ring[0][0]);
        __builtin_amdgcn_s_wait_tensorcnt((short)0);
    }
#endif
    __syncthreads();

    for (int t = 0; t < S; ++t) {
#if USE_TDM
        // Stream t+1's gate slice into the other ring slot; completion is
        // awaited before this step's final barrier, off the critical path.
        if (wid == 0 && t + 1 < S)
            tdm_issue_gate_slice(gate_base + (size_t)(t + 1) * 3072,
                                 &ring[(t + 1) & 1][0]);
#endif
        // ---- forget & reset gate matmuls: A from h_bf -------------------
        AccF accF, accR;
        #pragma unroll
        for (int i = 0; i < 8; ++i) { accF.f[i] = 0.f; accR.f[i] = 0.f; }
        #pragma unroll
        for (int c = 0; c < 4; ++c) {
            Frag16 a;
            #pragma unroll
            for (int v = 0; v < 8; ++v) {
                const int k = c * 32 + ((v < 4) ? (ks + 2 * v) : (16 + ks + 2 * (v - 4)));
                a.u[v] = (lm < B) ? *(const unsigned*)&h_bf[lm][k] : 0u;
            }
            accF.v = __builtin_amdgcn_wmma_f32_16x16x32_bf16(
                false, a.v, false, fWf[c].v, (short)0, accF.v, false, false);
            accR.v = __builtin_amdgcn_wmma_f32_16x16x32_bf16(
                false, a.v, false, fWr[c].v, (short)0, accR.v, false, false);
        }

        // ---- gate inputs ------------------------------------------------
        float iin[B] = {0.f, 0.f, 0.f, 0.f};
        float fin[B] = {0.f, 0.f, 0.f, 0.f};
        float rin[B] = {0.f, 0.f, 0.f, 0.f};
        if (act) {
            #pragma unroll
            for (int b = 0; b < B; ++b) {
#if USE_TDM
                const float* rp = &ring[t & 1][b * 3 * H + ncol];
                iin[b] = rp[0 * H];
                fin[b] = rp[1 * H];
                rin[b] = rp[2 * H];
#else
                const size_t base = (size_t)b * pstride_b + (size_t)t * 3072 + doff;
                iin[b] = proj[base];
                fin[b] = proj[base + 1024];
                rin[b] = proj[base + 2048];
                if (t + 1 < S) __builtin_prefetch(&proj[base + 3072], 0, 1);
#endif
            }
        }

        float fg[B] = {0.f, 0.f, 0.f, 0.f};
        if (act) {
            #pragma unroll
            for (int b = 0; b < B; ++b) {
                fg[b]          = sigmoidf_fast(accF.f[b] + fin[b]);
                const float rg = sigmoidf_fast(accR.f[b] + rin[b]);
                hr_bf[b][ncol] = (__bf16)(hreg[b] * rg);
            }
        }
        __syncthreads();   // hr_bf ready; all h_bf / ring reads of step t done

        // ---- candidate matmul: A from hr_bf -----------------------------
        AccF accC;
        #pragma unroll
        for (int i = 0; i < 8; ++i) accC.f[i] = 0.f;
        #pragma unroll
        for (int c = 0; c < 4; ++c) {
            Frag16 a;
            #pragma unroll
            for (int v = 0; v < 8; ++v) {
                const int k = c * 32 + ((v < 4) ? (ks + 2 * v) : (16 + ks + 2 * (v - 4)));
                a.u[v] = (lm < B) ? *(const unsigned*)&hr_bf[lm][k] : 0u;
            }
            accC.v = __builtin_amdgcn_wmma_f32_16x16x32_bf16(
                false, a.v, false, fWc[c].v, (short)0, accC.v, false, false);
        }

        if (act) {
            #pragma unroll
            for (int b = 0; b < B; ++b) {
                const float e2   = __expf(2.0f * (accC.f[b] + iin[b]));
                const float cand = (e2 - 1.0f) / (e2 + 1.0f);      // tanh
                const float hn   = fg[b] * hreg[b] + (1.0f - fg[b]) * cand;
                hreg[b] = hn;
                h_bf[b][ncol] = (__bf16)hn;
                states[((size_t)b * S + t) * 1024 + doff] = hn;
                if (t == S - 1) h_last[(size_t)b * 1024 + doff] = hn;
            }
        }
#if USE_TDM
        if (wid == 0) __builtin_amdgcn_s_wait_tensorcnt((short)0);
#endif
        __syncthreads();   // h_bf updated + next ring slot published
    }
    (void)pstride_b;
}

// ---------------------------------------------------------------------------
extern "C" void kernel_launch(void* const* d_in, const int* in_sizes, int n_in,
                              void* d_out, int out_size, void* d_ws, size_t ws_size,
                              hipStream_t stream) {
    (void)in_sizes; (void)n_in; (void)out_size; (void)ws_size;

    const float* x    = (const float*)d_in[0];   // (4,2048,1024)
    const float* Wp   = (const float*)d_in[1];   // (3072,1024)
    const float* bp   = (const float*)d_in[2];   // (3072,)
    const float* Wc   = (const float*)d_in[3];   // state_weight  (8,128,128)
    const float* Wf   = (const float*)d_in[4];   // forget_weight (8,128,128)
    const float* Wr   = (const float*)d_in[5];   // reset_weight  (8,128,128)
    const float* Wout = (const float*)d_in[6];   // (1024,1024)

    const int M = 4 * 2048;                      // B*S rows
    float* proj   = (float*)d_ws;                                        // 96 MB
    float* states = (float*)((char*)d_ws + (size_t)M * 3072 * sizeof(float)); // 32 MB

    float* out    = (float*)d_out;                        // (4,2048,1024)
    float* h_last = (float*)d_out + (size_t)M * 1024;     // (4,1024)

    // 1) proj = x @ Wp.T + bp
    gemm_bf16_wmma<<<dim3(3072 / 128, M / 128), 256, 0, stream>>>(
        x, Wp, bp, proj, M, 1024, 3072);

    // 2) recurrent scan (one block per head)
    gru_scan_wmma<<<dim3(8), 256, 0, stream>>>(proj, Wc, Wf, Wr, states, h_last);

    // 3) out = states @ Wout.T
    gemm_bf16_wmma<<<dim3(1024 / 128, M / 128), 256, 0, stream>>>(
        states, Wout, nullptr, out, M, 1024, 1024);
}
